// LSTMPeepholeBlock_57114475102581
// MI455X (gfx1250) — compile-verified, matrix operand order checked by reading, not run on hardware
//
#include <hip/hip_runtime.h>
#include <hip/hip_bf16.h>
#include <math.h>

typedef __attribute__((ext_vector_type(16))) __bf16 bf16x16;
typedef __attribute__((ext_vector_type(8)))  float  f32x8;

static constexpr int BB  = 4096;   // batch
static constexpr int DIN = 2048;   // input dim (== K for x-GEMMs)
static constexpr int NN  = 2048;   // hidden dim

// ---------------------------------------------------------------------------
// fp32 -> bf16 row-major convert
// ---------------------------------------------------------------------------
__global__ void f32_to_bf16_kernel(const float* __restrict__ in,
                                   __bf16* __restrict__ out, int n) {
    int i = (blockIdx.x * blockDim.x + threadIdx.x) * 4;
    if (i >= n) return;
    float4 v = *reinterpret_cast<const float4*>(in + i);
    union { __bf16 b[4]; uint2 u; } r;
    r.b[0] = (__bf16)v.x; r.b[1] = (__bf16)v.y;
    r.b[2] = (__bf16)v.z; r.b[3] = (__bf16)v.w;
    *reinterpret_cast<uint2*>(out + i) = r.u;
}

// ---------------------------------------------------------------------------
// fp32 [R x C] -> bf16 [C x R] transpose via LDS tile
// ---------------------------------------------------------------------------
__global__ void transpose_f32_to_bf16_kernel(const float* __restrict__ in,
                                             __bf16* __restrict__ out,
                                             int R, int C) {
    __shared__ __bf16 tile[32][34];
    const int c0 = blockIdx.x * 32, r0 = blockIdx.y * 32;
    const int tx = threadIdx.x, ty = threadIdx.y;   // block = (32, 8)
    #pragma unroll
    for (int j = 0; j < 32; j += 8)
        tile[ty + j][tx] = (__bf16)in[(size_t)(r0 + ty + j) * C + (c0 + tx)];
    __syncthreads();
    #pragma unroll
    for (int j = 0; j < 32; j += 8)
        out[(size_t)(c0 + ty + j) * R + (r0 + tx)] = tile[tx][ty + j];
}

// ---------------------------------------------------------------------------
// fragment loader: rows of a bf16 row-major [rows x K] matrix.
// lane&15 picks the row, lane>>4 picks the K-phase (CDNA5 16-bit A/B layout:
// v0..3 hold K=0..7 / 8..15, v4..7 hold K=16..23 / 24..31).
// ---------------------------------------------------------------------------
__device__ __forceinline__ bf16x16 load_frag(const __bf16* __restrict__ p) {
    union { uint4 u[2]; bf16x16 v; } f;
    f.u[0] = *reinterpret_cast<const uint4*>(p);
    f.u[1] = *reinterpret_cast<const uint4*>(p + 16);
    return f.v;
}

__device__ __forceinline__ f32x8 wmma_bf16(bf16x16 a, bf16x16 b, f32x8 c) {
    return __builtin_amdgcn_wmma_f32_16x16x32_bf16(
        false, a, false, b, (short)0, c, false, false);
}

__device__ __forceinline__ float sigmoidf_(float x) {
    return 1.0f / (1.0f + __expf(-x));
}

// ---------------------------------------------------------------------------
// Fused LSTM, register-blocked: each wave computes a 32(M) x 16(N) tile for
// each of 7 GEMMs. Two M-subtiles share every B fragment -> 1.5 b128 loads
// per WMMA (vs 2.5 unblocked). Workgroup = 8 waves (2M x 4N) -> 64x64 tile.
//   acc[g][s]: g in {xf,xi,xo,xc,hf,hi,ho}, s = M-subtile (m0 / m0+16)
// ---------------------------------------------------------------------------
__global__ __launch_bounds__(256)
void lstm_fused_wmma_kernel(const __bf16* __restrict__ xbf,
                            const __bf16* __restrict__ hbf,
                            const __bf16* __restrict__ Wt,   // [4][NN][DIN] bf16 (transposed)
                            const __bf16* __restrict__ Ut,   // [3][NN][NN]  bf16 (transposed)
                            const float*  __restrict__ c_in,
                            const float*  __restrict__ bf_v,
                            const float*  __restrict__ bi_v,
                            const float*  __restrict__ bo_v,
                            const float*  __restrict__ bc_v,
                            float* __restrict__ out_h,
                            float* __restrict__ out_c) {
    const int lane = threadIdx.x & 31;
    const int wave = threadIdx.x >> 5;
    const int m0 = blockIdx.y * 64 + (wave >> 2) * 32;   // wave owns rows m0..m0+31
    const int n0 = blockIdx.x * 64 + (wave & 3) * 16;    // wave owns cols n0..n0+15

    const int arow   = m0 + (lane & 15);
    const int brow   = n0 + (lane & 15);
    const int kphase = (lane >> 4) << 3;   // 0 or 8

    f32x8 acc[7][2];
    #pragma unroll
    for (int g = 0; g < 7; ++g)
        #pragma unroll
        for (int s = 0; s < 2; ++s)
            acc[g][s] = (f32x8){0.f, 0.f, 0.f, 0.f, 0.f, 0.f, 0.f, 0.f};

    // ---- phase 1: x @ W{f,i,o,c}, K = DIN ----
    {
        const __bf16* aP = xbf + (size_t)arow * DIN + kphase;
        const __bf16* bP = Wt  + (size_t)brow * DIN + kphase;
        #pragma unroll 2
        for (int k = 0; k < DIN; k += 32) {
            bf16x16 a0 = load_frag(aP + k);
            bf16x16 a1 = load_frag(aP + (size_t)16 * DIN + k);
            #pragma unroll
            for (int g = 0; g < 4; ++g) {
                bf16x16 b = load_frag(bP + (size_t)g * NN * DIN + k);
                acc[g][0] = wmma_bf16(a0, b, acc[g][0]);
                acc[g][1] = wmma_bf16(a1, b, acc[g][1]);
            }
        }
    }
    // ---- phase 2: h @ U{f,i,o}, K = NN ----
    {
        const __bf16* aP = hbf + (size_t)arow * NN + kphase;
        const __bf16* bP = Ut  + (size_t)brow * NN + kphase;
        #pragma unroll 2
        for (int k = 0; k < NN; k += 32) {
            bf16x16 a0 = load_frag(aP + k);
            bf16x16 a1 = load_frag(aP + (size_t)16 * NN + k);
            #pragma unroll
            for (int g = 0; g < 3; ++g) {
                bf16x16 b = load_frag(bP + (size_t)g * NN * NN + k);
                acc[4 + g][0] = wmma_bf16(a0, b, acc[4 + g][0]);
                acc[4 + g][1] = wmma_bf16(a1, b, acc[4 + g][1]);
            }
        }
    }

    // ---- epilogue: gates + state update ----
    // C/D layout: VGPR e -> row mBase+e, mBase = m0(+16*s) + 8*(lane>=16);
    //             column = n0 + (lane & 15).
    const int nn  = n0 + (lane & 15);
    const float vbf = bf_v[nn], vbi = bi_v[nn], vbo = bo_v[nn], vbc = bc_v[nn];
    #pragma unroll
    for (int s = 0; s < 2; ++s) {
        const int mBase = m0 + 16 * s + ((lane & 16) ? 8 : 0);
        #pragma unroll
        for (int e = 0; e < 8; ++e) {
            const size_t idx = (size_t)(mBase + e) * NN + nn;
            const float fg = sigmoidf_(acc[0][s][e] + acc[4][s][e] + vbf);
            const float ig = sigmoidf_(acc[1][s][e] + acc[5][s][e] + vbi);
            const float og = sigmoidf_(acc[2][s][e] + acc[6][s][e] + vbo);
            const float cn = fg * c_in[idx] + ig * tanhf(acc[3][s][e] + vbc);
            out_h[idx] = og * cn;
            out_c[idx] = cn;
        }
    }
}

// ---------------------------------------------------------------------------
// launch
// ---------------------------------------------------------------------------
extern "C" void kernel_launch(void* const* d_in, const int* in_sizes, int n_in,
                              void* d_out, int out_size, void* d_ws, size_t ws_size,
                              hipStream_t stream) {
    // input order: x h c Wf Wi Wo Wc bc Uf bf Ui bi Uo bo
    const float* x  = (const float*)d_in[0];
    const float* h  = (const float*)d_in[1];
    const float* c  = (const float*)d_in[2];
    const float* W[4] = { (const float*)d_in[3], (const float*)d_in[4],
                          (const float*)d_in[5], (const float*)d_in[6] }; // f,i,o,c
    const float* bc = (const float*)d_in[7];
    const float* U[3] = { (const float*)d_in[8], (const float*)d_in[10],
                          (const float*)d_in[12] };                       // f,i,o
    const float* bf = (const float*)d_in[9];
    const float* bi = (const float*)d_in[11];
    const float* bo = (const float*)d_in[13];

    // workspace layout (bf16): xbf | hbf | Wt[4] | Ut[3]   (~88 MB)
    __bf16* ws  = (__bf16*)d_ws;
    __bf16* xbf = ws;                                   // BB*DIN
    __bf16* hbf = xbf + (size_t)BB * DIN;               // BB*NN
    __bf16* Wt  = hbf + (size_t)BB * NN;                // 4 * NN*DIN
    __bf16* Ut  = Wt  + (size_t)4 * NN * DIN;           // 3 * NN*NN

    // --- convert activations to bf16 ---
    {
        const int n = BB * DIN;
        f32_to_bf16_kernel<<<n / (256 * 4), 256, 0, stream>>>(x, xbf, n);
    }
    {
        const int n = BB * NN;
        f32_to_bf16_kernel<<<n / (256 * 4), 256, 0, stream>>>(h, hbf, n);
    }

    // --- transpose + convert weights: [K x N] fp32 -> [N x K] bf16 ---
    dim3 tb(32, 8);
    dim3 tgW(NN / 32, DIN / 32);
    for (int g = 0; g < 4; ++g)
        transpose_f32_to_bf16_kernel<<<tgW, tb, 0, stream>>>(
            W[g], Wt + (size_t)g * NN * DIN, DIN, NN);
    dim3 tgU(NN / 32, NN / 32);
    for (int g = 0; g < 3; ++g)
        transpose_f32_to_bf16_kernel<<<tgU, tb, 0, stream>>>(
            U[g], Ut + (size_t)g * NN * NN, NN, NN);

    // --- fused LSTM GEMM + gates ---
    float* out_h = (float*)d_out;
    float* out_c = out_h + (size_t)BB * NN;
    dim3 grid(NN / 64, BB / 64);
    lstm_fused_wmma_kernel<<<grid, 256, 0, stream>>>(
        xbf, hbf, Wt, Ut, c, bf, bi, bo, bc, out_h, out_c);
}
